// PreInitMLP_65000035057932
// MI455X (gfx1250) — matrix operand order repaired and checked by reading, not compiled, not run on hardware
//
#include <hip/hip_runtime.h>
#include <hip/hip_bf16.h>

typedef __attribute__((ext_vector_type(16))) _Float16 v16h;
typedef __attribute__((ext_vector_type(8)))  float    v8f;

#define SATC 10.0f
#define NLOG2E   1.44269504088896340736f          // log2(e)
#define NSLOG2E  14.4269504088896340736f          // SATC * log2(e)

// sigmoid(SATC * pre): fast path -> v_mul, v_exp_f32, v_add, v_rcp_f32
__device__ __forceinline__ float sig10(float pre) {
    const float e = __builtin_amdgcn_exp2f(pre * (-NSLOG2E));
    return __builtin_amdgcn_rcpf(1.0f + e);
}

// plain sigmoid(x) (used for non-last stage output squash)
__device__ __forceinline__ float sigf(float x) {
    const float e = __builtin_amdgcn_exp2f(x * (-NLOG2E));
    return __builtin_amdgcn_rcpf(1.0f + e);
}

// ---------------------------------------------------------------------------
// First layer as WMMA: G = p (1024x32) @ W (32x2048), W[m, i*64+j] = (j-i==m).
// Exact in f16 since all operands are 0/1 and each output has <=1 nonzero term.
// One wave per (16-row batch tile, 16-col tile): 64 * 128 = 8192 waves.
// ---------------------------------------------------------------------------
__global__ __launch_bounds__(256) void first_layer_wmma(const float* __restrict__ x,
                                                        float* __restrict__ out)
{
    const int lane = threadIdx.x & 31;
    const int wave = blockIdx.x * (blockDim.x >> 5) + (threadIdx.x >> 5);
    const int bt = wave >> 7;        // batch tile 0..63
    const int ct = wave & 127;       // column tile 0..127
    const int half = lane >> 4;      // 0: lanes 0-15, 1: lanes 16-31
    const int l16  = lane & 15;

    // --- A fragment: 16x32 f16, documented layout:
    // lanes 0-15 : halves 0..7 = K 0..7,  halves 8..15 = K 16..23
    // lanes 16-31: halves 0..7 = K 8..15, halves 8..15 = K 24..31
    const int arow  = bt * 16 + l16;
    const int kbase = half * 8;
    const float* prow = x + (size_t)arow * 64;
    v16h a;
    #pragma unroll
    for (int h = 0; h < 8; ++h) a[h]     = (_Float16)prow[kbase + h];
    #pragma unroll
    for (int h = 0; h < 8; ++h) a[8 + h] = (_Float16)prow[kbase + 16 + h];

    // --- B fragment: 32x16 f16 shift matrix, layout:
    // lane L holds column N=L&15; lanes 0-15 rows K=0..7,16..23, lanes 16-31 rows K=8..15,24..31
    const int i_col = ct >> 2;            // i index (0..31), 4 col-tiles per i
    const int j0    = (ct & 3) * 16;      // j base within this i
    const int jn    = j0 + l16;           // this lane's j
    const int mbase = half * 8;
    v16h w;
    #pragma unroll
    for (int h = 0; h < 16; ++h) {
        w[h] = (_Float16)((jn - i_col == mbase + h + ((h >= 8) ? 8 : 0)) ? 1.0f : 0.0f);
    }

    v8f c = {};
    c = __builtin_amdgcn_wmma_f32_16x16x32_f16(
        /*neg_a=*/false, a, /*neg_b=*/false, w,
        /*c_mod=*/(short)0, c, /*reuse_a=*/false, /*reuse_b=*/false);

    // --- Epilogue: pre = valid ? 2*(pg + q) - 3 : -1 ; out = sigmoid(10*pre)
    // Unconditional q load + select (no divergent exec regions).
    const int ji     = jn - i_col;
    const bool valid = (ji >= 0) && (ji < 32);
    #pragma unroll
    for (int r = 0; r < 8; ++r) {
        const int brow = bt * 16 + r + half * 8;   // C/D layout: M = r (+8 for hi lanes)
        const float qv = x[(size_t)brow * 64 + 32 + i_col];
        float pre = 2.0f * (c[r] + qv) - 3.0f;
        pre = valid ? pre : -1.0f;
        out[(size_t)brow * 2048 + i_col * 64 + jn] = sig10(pre);
    }
}

// ---------------------------------------------------------------------------
// Addition stage: one wave per (batch, pair). Lane L owns j = {2L, 2L+1},
// t[2][3] lives in registers; 64-step scan uses wave shuffles for t[i-1][*].
// ---------------------------------------------------------------------------
template<int LAST>
__global__ __launch_bounds__(256) void add_stage(const float* __restrict__ in,
                                                 float* __restrict__ out,
                                                 int BP, int P)
{
    const int lane = threadIdx.x & 31;
    const int wave = blockIdx.x * (blockDim.x >> 5) + (threadIdx.x >> 5);
    if (wave >= BP) return;
    const int b = wave / P;
    const int p = wave % P;
    const float* arow = in + ((size_t)b * (2 * P) + 2 * p) * 64;
    const float* brow = arow + 64;

    float t[2][3];
    #pragma unroll
    for (int m = 0; m < 2; ++m) {
        const int j = 2 * lane + m;
        const float apb = arow[j] + brow[j];
        t[m][0] = sig10(2.0f * apb - 3.0f);
        t[m][1] = sig10(2.0f * apb - 1.0f);
        t[m][2] = sig10(1.0f - 2.0f * apb);
    }

    // Sequential carry scan: at step i, read t[i-1][0], t[i-1][1] (pre-update),
    // sharpen every element with sigmoid(s*(2t-1)), override slots [i][1],[i][2].
    for (int i = 0; i < 64; ++i) {
        const int pl = (i > 0) ? ((i - 1) >> 1) : 0;
        const float c0 = __shfl(t[0][0], pl);
        const float c1 = __shfl(t[1][0], pl);
        const float q0 = __shfl(t[0][1], pl);
        const float q1 = __shfl(t[1][1], pl);
        const bool odd = ((i - 1) & 1) != 0;
        const float c_prev = odd ? c1 : c0;
        const float p_prev = odd ? q1 : q0;
        #pragma unroll
        for (int m = 0; m < 2; ++m) {
            const int j = 2 * lane + m;
            float pre0 = 2.0f * t[m][0] - 1.0f;
            float pre1 = 2.0f * t[m][1] - 1.0f;
            float pre2 = 2.0f * t[m][2] - 1.0f;
            if (j == i) {
                pre1 = (i == 0) ? -1.0f
                                : (2.0f * (c_prev + p_prev) - 5.0f + 4.0f * t[m][1]);
                pre2 = 1.0f - 2.0f * (t[m][0] + t[m][2]);
            }
            t[m][0] = sig10(pre0);
            t[m][1] = sig10(pre1);
            t[m][2] = sig10(pre2);
        }
    }

    // Final combine: carry_prev[j] = (j==0) ? -1 : -1 + 2*(t[j-1][0]+t[j-1][1])
    const float n0 = __shfl_up(t[1][0], 1);
    const float n1 = __shfl_up(t[1][1], 1);
    float* orow = out + ((size_t)b * P + p) * 64;
    #pragma unroll
    for (int m = 0; m < 2; ++m) {
        const int j = 2 * lane + m;
        float cp;
        if (m == 1) cp = -1.0f + 2.0f * (t[0][0] + t[0][1]);
        else        cp = (j == 0) ? -1.0f : (-1.0f + 2.0f * (n0 + n1));
        const float xo = t[m][2];
        const float u1 = sig10( 1.0f - 4.0f * xo - 2.0f * cp);
        const float u2 = sig10(-7.0f + 6.0f * xo + 2.0f * cp);
        float o = SATC * (1.0f - 2.0f * (u1 + u2));
        if (!LAST) o = sigf(o);
        orow[j] = o;
    }
}

extern "C" void kernel_launch(void* const* d_in, const int* in_sizes, int n_in,
                              void* d_out, int out_size, void* d_ws, size_t ws_size,
                              hipStream_t stream) {
    (void)in_sizes; (void)n_in; (void)out_size; (void)ws_size;
    const float* x = (const float*)d_in[0];
    const int B = 1024;

    float* bufA = (float*)d_ws;                                        // 8 MB region
    float* bufB = (float*)((char*)d_ws + (size_t)B * 32 * 64 * 4);     // 4 MB region

    // First layer -> bufA  [B][32][64]   (8192 waves = 1024 blocks x 8 waves)
    first_layer_wmma<<<1024, 256, 0, stream>>>(x, bufA);

    // Stages P = 16, 8, 4, 2 (ping-pong), then last P = 1 -> d_out
    const float* src = bufA;
    float* dst = bufB;
    int P = 16;
    for (int d = 0; d < 4; ++d) {
        const int BP = B * P;
        add_stage<0><<<(BP + 7) / 8, 256, 0, stream>>>(src, dst, BP, P);
        src = dst;
        dst = (dst == bufB) ? bufA : bufB;
        P >>= 1;
    }
    add_stage<1><<<(B + 7) / 8, 256, 0, stream>>>(src, (float*)d_out, B, 1);
}
